// GPT_10651518894803
// MI455X (gfx1250) — compile-verified, matrix operand order checked by reading, not run on hardware
//
#include <hip/hip_runtime.h>
#include <hip/hip_bf16.h>
#include <math.h>

typedef __bf16 bf16_t;
typedef bf16_t v16bf __attribute__((ext_vector_type(16)));
typedef bf16_t v8bf  __attribute__((ext_vector_type(8)));
typedef bf16_t v4bf  __attribute__((ext_vector_type(4)));
typedef float  v8f   __attribute__((ext_vector_type(8)));
typedef float  v4f   __attribute__((ext_vector_type(4)));

#define GPT_B   16
#define GPT_S   128
#define GPT_L   12
#define GPT_H   12
#define GPT_D   768
#define GPT_DH  64
#define GPT_HID 3072
#define GPT_V   40000
#define GPT_T   (GPT_B * GPT_S)   // 2048 tokens (M for every GEMM)

// ---------------------------------------------------------------------------
// Embedding: h[b,s,:] = bpe[x[b,s],:] + pos[s,:]
// ---------------------------------------------------------------------------
__global__ __launch_bounds__(256) void k_embed(const int* __restrict__ x,
                                               const float* __restrict__ bpe,
                                               const float* __restrict__ pos,
                                               float* __restrict__ h) {
    int idx = blockIdx.x * 256 + threadIdx.x;
    if (idx >= GPT_T * GPT_D) return;
    int d = idx % GPT_D;
    int t = idx / GPT_D;
    int s = t % GPT_S;
    h[idx] = bpe[(size_t)x[t] * GPT_D + d] + pos[s * GPT_D + d];
}

// ---------------------------------------------------------------------------
// LayerNorm over D=768, one workgroup (256 threads) per token row
// ---------------------------------------------------------------------------
__global__ __launch_bounds__(256) void k_layernorm(const float* __restrict__ X,
                                                   const float* __restrict__ g,
                                                   const float* __restrict__ b,
                                                   float* __restrict__ Y) {
    int row = blockIdx.x;
    const float* xr = X + (size_t)row * GPT_D;
    float*       yr = Y + (size_t)row * GPT_D;
    int t = threadIdx.x;
    float x0 = xr[t], x1 = xr[t + 256], x2 = xr[t + 512];

    __shared__ float red[256];
    red[t] = x0 + x1 + x2;
    __syncthreads();
    for (int off = 128; off > 0; off >>= 1) {
        if (t < off) red[t] += red[t + off];
        __syncthreads();
    }
    float mean = red[0] * (1.0f / GPT_D);
    __syncthreads();
    float d0 = x0 - mean, d1 = x1 - mean, d2 = x2 - mean;
    red[t] = d0 * d0 + d1 * d1 + d2 * d2;
    __syncthreads();
    for (int off = 128; off > 0; off >>= 1) {
        if (t < off) red[t] += red[t + off];
        __syncthreads();
    }
    float var  = red[0] * (1.0f / GPT_D);
    float rstd = rsqrtf(var + 1e-5f);
    yr[t]       = d0 * rstd * g[t]       + b[t];
    yr[t + 256] = d1 * rstd * g[t + 256] + b[t + 256];
    yr[t + 512] = d2 * rstd * g[t + 512] + b[t + 512];
}

// ---------------------------------------------------------------------------
// Attention: one workgroup of 128 threads per (batch, head); thread = query row.
// keep = (k<=q) && (ignore[b,k]==0 || k==q). Two-pass softmax; K staged in LDS.
// ---------------------------------------------------------------------------
__global__ __launch_bounds__(128) void k_attention(const float* __restrict__ q,
                                                   const float* __restrict__ k,
                                                   const float* __restrict__ v,
                                                   const int* __restrict__ ign,
                                                   float* __restrict__ o) {
    int bh   = blockIdx.x;
    int b    = bh / GPT_H;
    int head = bh % GPT_H;
    size_t base = (size_t)b * GPT_S * GPT_D + (size_t)head * GPT_DH;

    __shared__ float ks[GPT_S][GPT_DH];   // 32 KB
    __shared__ int   ig[GPT_S];
    int tid = threadIdx.x;
    #pragma unroll
    for (int i = 0; i < (GPT_S * GPT_DH) / 128; ++i) {
        int idx = tid + i * 128;
        int ss = idx >> 6, e = idx & 63;
        ks[ss][e] = k[base + (size_t)ss * GPT_D + e];
    }
    ig[tid] = ign[b * GPT_S + tid];
    __syncthreads();

    int qi = tid;
    float qr[GPT_DH];
    #pragma unroll
    for (int e = 0; e < GPT_DH; ++e) qr[e] = q[base + (size_t)qi * GPT_D + e];

    float mx = -1e30f;
    for (int kk = 0; kk <= qi; ++kk) {
        if (ig[kk] != 0 && kk != qi) continue;
        float dot = 0.f;
        #pragma unroll
        for (int e = 0; e < GPT_DH; ++e) dot = fmaf(qr[e], ks[kk][e], dot);
        mx = fmaxf(mx, dot * 0.125f);
    }
    float acc[GPT_DH];
    #pragma unroll
    for (int e = 0; e < GPT_DH; ++e) acc[e] = 0.f;
    float sum = 0.f;
    const float* vbase = v + base;
    for (int kk = 0; kk <= qi; ++kk) {
        if (ig[kk] != 0 && kk != qi) continue;
        float dot = 0.f;
        #pragma unroll
        for (int e = 0; e < GPT_DH; ++e) dot = fmaf(qr[e], ks[kk][e], dot);
        float w = expf(dot * 0.125f - mx);
        sum += w;
        const float* vr = vbase + (size_t)kk * GPT_D;
        #pragma unroll
        for (int e = 0; e < GPT_DH; ++e) acc[e] = fmaf(w, vr[e], acc[e]);
    }
    float inv = 1.f / sum;
    #pragma unroll
    for (int e = 0; e < GPT_DH; ++e)
        o[base + (size_t)qi * GPT_D + e] = acc[e] * inv;
}

// ---------------------------------------------------------------------------
// bf16 WMMA GEMM: C[2048,N] = epilogue( A[2048,K] * W + bias ).
// 256 threads = 8 waves; block tile 128x128; K-chunk 32.
// Both A and B staged K-major per row/column (stride 40 bf16 = 80 B, 16B
// aligned) so every WMMA fragment is two ds_load_b128.
// MODE: 0 = +bias ; 1 = +bias+residual ; 2 = gelu(+bias)+residual.
// QKV: weight stored as (head, k, e) blocks, col = head*64 + e.
// ---------------------------------------------------------------------------
#define BM 128
#define BN 128
#define KC 32
#define KPAD 8   // row stride 40 bf16 = 80 B (multiple of 16 B)

template <int QKV, int MODE>
__global__ __launch_bounds__(256) void k_gemm_bf16(const float* __restrict__ A,
                                                   const float* __restrict__ W,
                                                   const float* __restrict__ bias,
                                                   const float* __restrict__ res,
                                                   float* __restrict__ C,
                                                   int N, int K) {
    __shared__ bf16_t As[BM][KC + KPAD];
    __shared__ bf16_t Bs[BN][KC + KPAD];  // transposed: column-major tile, K contiguous

    int tid  = threadIdx.x;
    int lane = tid & 31;
    int wave = tid >> 5;
    int wm   = (wave & 1) * 64;
    int wn   = (wave >> 1) * 32;
    int bm   = blockIdx.y * BM;
    int bn   = blockIdx.x * BN;
    bool full_n = (bn + BN <= N);

    v8f acc[4][2] = {};
    int mloc  = lane & 15;
    int khalf = (lane >> 4) << 3;   // 0 or 8 : K-half selected by lane group

    for (int k0 = 0; k0 < K; k0 += KC) {
        // ---- stage A tile: 128x32, float4 global loads -> bf16x4 LDS stores ----
        #pragma unroll
        for (int i = 0; i < 4; ++i) {
            int idx = tid + i * 256;           // 1024 quads
            int row = idx >> 3;
            int kq  = (idx & 7) << 2;
            v4f a = *(const v4f*)(A + (size_t)(bm + row) * K + (k0 + kq));
            v4bf p;
            p.x = (bf16_t)a.x; p.y = (bf16_t)a.y;
            p.z = (bf16_t)a.z; p.w = (bf16_t)a.w;
            *(v4bf*)&As[row][kq] = p;
        }
        // ---- stage B tile transposed: each thread loads 4 K-values of one col ----
        #pragma unroll
        for (int i = 0; i < 4; ++i) {
            int idx = tid + i * 256;           // 1024 quads
            int c   = idx & 127;
            int kq  = (idx >> 7) << 2;
            int gc  = bn + c;
            v4bf p;
            if (full_n) {
                size_t off0 = QKV
                    ? ((size_t)(gc >> 6) * ((size_t)K * 64) + (size_t)(gc & 63)
                       + (size_t)(k0 + kq) * 64)
                    : ((size_t)(k0 + kq) * (size_t)N + (size_t)gc);
                size_t str = QKV ? 64 : (size_t)N;
                p.x = (bf16_t)W[off0];
                p.y = (bf16_t)W[off0 + str];
                p.z = (bf16_t)W[off0 + 2 * str];
                p.w = (bf16_t)W[off0 + 3 * str];
            } else if (gc < N) {
                size_t off0 = QKV
                    ? ((size_t)(gc >> 6) * ((size_t)K * 64) + (size_t)(gc & 63)
                       + (size_t)(k0 + kq) * 64)
                    : ((size_t)(k0 + kq) * (size_t)N + (size_t)gc);
                size_t str = QKV ? 64 : (size_t)N;
                p.x = (bf16_t)W[off0];
                p.y = (bf16_t)W[off0 + str];
                p.z = (bf16_t)W[off0 + 2 * str];
                p.w = (bf16_t)W[off0 + 3 * str];
            } else {
                p.x = (bf16_t)0.f; p.y = (bf16_t)0.f;
                p.z = (bf16_t)0.f; p.w = (bf16_t)0.f;
            }
            *(v4bf*)&Bs[c][kq] = p;
        }
        __syncthreads();

        // ---- fragments: two 16-byte chunks each (K = khalf..+7, 16+khalf..+7) ----
        union FR { v16bf v; v8bf h[2]; };
        FR afrag[4], bfrag[2];
        #pragma unroll
        for (int tm = 0; tm < 4; ++tm) {
            int mrow = wm + tm * 16 + mloc;
            afrag[tm].h[0] = *(const v8bf*)&As[mrow][khalf];
            afrag[tm].h[1] = *(const v8bf*)&As[mrow][16 + khalf];
        }
        #pragma unroll
        for (int tn = 0; tn < 2; ++tn) {
            int ncol = wn + tn * 16 + mloc;
            bfrag[tn].h[0] = *(const v8bf*)&Bs[ncol][khalf];
            bfrag[tn].h[1] = *(const v8bf*)&Bs[ncol][16 + khalf];
        }
        // ---- 8 WMMAs per K-step ----
        #pragma unroll
        for (int tm = 0; tm < 4; ++tm)
            #pragma unroll
            for (int tn = 0; tn < 2; ++tn)
                acc[tm][tn] = __builtin_amdgcn_wmma_f32_16x16x32_bf16(
                    false, afrag[tm].v, false, bfrag[tn].v,
                    (short)0, acc[tm][tn], false, false);
        __syncthreads();
    }

    // ---- epilogue: C tile lane = N col, VGPR r = M row (+8 for upper lanes) ----
    int nlane = lane & 15;
    int mofs  = (lane >> 4) << 3;
    #pragma unroll
    for (int tm = 0; tm < 4; ++tm) {
        #pragma unroll
        for (int tn = 0; tn < 2; ++tn) {
            int col = bn + wn + tn * 16 + nlane;
            if (col >= N) continue;
            float bcol = bias[col];
            #pragma unroll
            for (int r = 0; r < 8; ++r) {
                int row = bm + wm + tm * 16 + mofs + r;   // always < 2048
                float xv = acc[tm][tn][r] + bcol;
                size_t oidx = (size_t)row * (size_t)N + (size_t)col;
                if (MODE == 1) {
                    xv += res[oidx];
                } else if (MODE == 2) {
                    // exact erf GELU applied AFTER second linear, then residual
                    xv = 0.5f * xv * (1.f + erff(xv * 0.70710678118654752f));
                    xv += res[oidx];
                }
                C[oidx] = xv;
            }
        }
    }
}

// ---------------------------------------------------------------------------
extern "C" void kernel_launch(void* const* d_in, const int* in_sizes, int n_in,
                              void* d_out, int out_size, void* d_ws, size_t ws_size,
                              hipStream_t stream) {
    (void)in_sizes; (void)n_in; (void)out_size; (void)ws_size;
    const int*   x    = (const int*)d_in[0];
    const int*   ign  = (const int*)d_in[1];
    const float* bpe  = (const float*)d_in[2];
    const float* pos  = (const float*)d_in[3];
    const float* Wq   = (const float*)d_in[4];
    const float* bq   = (const float*)d_in[5];
    const float* Wk   = (const float*)d_in[6];
    const float* bk   = (const float*)d_in[7];
    const float* Wv   = (const float*)d_in[8];
    const float* bv   = (const float*)d_in[9];
    const float* Wo   = (const float*)d_in[10];
    const float* bo   = (const float*)d_in[11];
    const float* W1   = (const float*)d_in[12];
    const float* b1   = (const float*)d_in[13];
    const float* W2   = (const float*)d_in[14];
    const float* b2   = (const float*)d_in[15];
    const float* g1   = (const float*)d_in[16];
    const float* be1  = (const float*)d_in[17];
    const float* g2   = (const float*)d_in[18];
    const float* be2  = (const float*)d_in[19];
    const float* Wout = (const float*)d_in[20];
    const float* bout = (const float*)d_in[21];

    float* ws = (float*)d_ws;
    const size_t TD = (size_t)GPT_T * GPT_D;
    float* h  = ws;            // [T, D]
    float* qb = h  + TD;       // [T, D]
    float* kb = qb + TD;       // [T, D]
    float* vb = kb + TD;       // [T, D]
    float* ob = vb + TD;       // [T, D]
    float* sm = ob + TD;       // [T, D] pre-LN residual sums
    float* ub = sm + TD;       // [T, HID]

    dim3 blk(256);
    dim3 g768(GPT_D / BN, GPT_T / BM);             // (6, 16)
    dim3 g3072(GPT_HID / BN, GPT_T / BM);          // (24, 16)
    dim3 gout((GPT_V + BN - 1) / BN, GPT_T / BM);  // (313, 16)

    k_embed<<<dim3((GPT_T * GPT_D + 255) / 256), blk, 0, stream>>>(x, bpe, pos, h);

    const size_t WQKV = (size_t)GPT_H * GPT_D * GPT_DH;
    const size_t WOO  = (size_t)GPT_D * GPT_D;
    const size_t WF1  = (size_t)GPT_D * GPT_HID;
    const size_t WF2  = (size_t)GPT_HID * GPT_D;

    for (int l = 0; l < GPT_L; ++l) {
        // QKV projections (head-blocked weights)
        k_gemm_bf16<1, 0><<<g768, blk, 0, stream>>>(h, Wq + (size_t)l * WQKV,
            bq + (size_t)l * GPT_D, nullptr, qb, GPT_D, GPT_D);
        k_gemm_bf16<1, 0><<<g768, blk, 0, stream>>>(h, Wk + (size_t)l * WQKV,
            bk + (size_t)l * GPT_D, nullptr, kb, GPT_D, GPT_D);
        k_gemm_bf16<1, 0><<<g768, blk, 0, stream>>>(h, Wv + (size_t)l * WQKV,
            bv + (size_t)l * GPT_D, nullptr, vb, GPT_D, GPT_D);

        // attention per (batch, head)
        k_attention<<<GPT_B * GPT_H, 128, 0, stream>>>(qb, kb, vb, ign, ob);

        // output projection + residual, then LN -> h
        k_gemm_bf16<0, 1><<<g768, blk, 0, stream>>>(ob, Wo + (size_t)l * WOO,
            bo + (size_t)l * GPT_D, h, sm, GPT_D, GPT_D);
        k_layernorm<<<GPT_T, 256, 0, stream>>>(sm, g1 + (size_t)l * GPT_D,
            be1 + (size_t)l * GPT_D, h);

        // FFN: u = h@W1 + b1 ; f = gelu(u@W2 + b2) ; h = LN(h + f)
        k_gemm_bf16<0, 0><<<g3072, blk, 0, stream>>>(h, W1 + (size_t)l * WF1,
            b1 + (size_t)l * GPT_HID, nullptr, ub, GPT_HID, GPT_D);
        k_gemm_bf16<0, 2><<<g768, blk, 0, stream>>>(ub, W2 + (size_t)l * WF2,
            b2 + (size_t)l * GPT_D, h, sm, GPT_D, GPT_HID);
        k_layernorm<<<GPT_T, 256, 0, stream>>>(sm, g2 + (size_t)l * GPT_D,
            be2 + (size_t)l * GPT_D, h);
    }

    // logits: [2048 x 768] x [768 x 40000]
    k_gemm_bf16<0, 0><<<gout, blk, 0, stream>>>(h, Wout, bout, nullptr,
        (float*)d_out, GPT_V, GPT_D);
}
